// WeightedCausalMultiheadAttention_41008347742353
// MI455X (gfx1250) — matrix operand order repaired and checked by reading, not compiled, not run on hardware
//
#include <hip/hip_runtime.h>
#include <hip/hip_bf16.h>

typedef __attribute__((ext_vector_type(16))) _Float16 v16h;
typedef __attribute__((ext_vector_type(4)))  _Float16 v4h;
typedef __attribute__((ext_vector_type(8)))  float    v8f;
typedef __attribute__((ext_vector_type(4)))  int      v4i;

#define B_  2
#define T_  2048
#define D_  1024
#define H_  16
#define HD_ 64
#define M_  (B_*T_)

#if __has_builtin(__builtin_amdgcn_global_load_async_to_lds_b128)
#define HAVE_ASYNC_LDS 1
// builtin signature (from hipcc diagnostic): (v4i AS1*, v4i AS3*, imm, imm)
#define ASYNC_CP_B128(g, l)                                                  \
  __builtin_amdgcn_global_load_async_to_lds_b128(                            \
      (__attribute__((address_space(1))) v4i*)(v4i*)(void*)(g),              \
      (__attribute__((address_space(3))) v4i*)(v4i*)(void*)(l), 0, 0)
#endif

#if __has_builtin(__builtin_amdgcn_s_wait_asynccnt)
#define WAIT_ASYNC() __builtin_amdgcn_s_wait_asynccnt(0)
#else
#define WAIT_ASYNC() asm volatile("s_wait_asynccnt 0" ::: "memory")
#endif

// ---------------- WMMA fragment helpers (CDNA5 wave32, 16x16x32 f16) -------
// A fragment 16x32 (MxK), memory row-major src[m*ld + k].
// Layout: lane L holds row L%16; lanes<16: K 0..15,
// lanes>=16: K 16..31.
__device__ __forceinline__ v16h frag_a_rowmajor(const _Float16* src, int ld) {
  const int lane = threadIdx.x & 31;
  const int row  = lane & 15;
  const int kb   = (lane < 16) ? 0 : 16;
  const _Float16* p = src + (size_t)row * ld + kb;
  v16h a;
#pragma unroll
  for (int i = 0; i < 16; ++i) a[i] = p[i];
  return a;
}

// B fragment 32x16 (KxN), memory N-major src[n*ld + k].
// Layout: lane holds column N=lane%16; lanes<16: K 0..15, lanes>=16: K 16..31.
__device__ __forceinline__ v16h frag_b_nmajor(const _Float16* src, int ld) {
  const int lane = threadIdx.x & 31;
  const int col  = lane & 15;
  const int kb   = (lane < 16) ? 0 : 16;
  const _Float16* p = src + (size_t)col * ld + kb;
  v16h b;
#pragma unroll
  for (int i = 0; i < 16; ++i) b[i] = p[i];
  return b;
}

// ---------------- fp32 -> f16 conversion ------------------------------------
__global__ void cvt_f32_f16(const float* __restrict__ s, _Float16* __restrict__ d, int n4) {
  int i = blockIdx.x * blockDim.x + threadIdx.x;
  if (i >= n4) return;
  float4 f = ((const float4*)s)[i];
  v4h h;
  h[0] = (_Float16)f.x; h[1] = (_Float16)f.y;
  h[2] = (_Float16)f.z; h[3] = (_Float16)f.w;
  ((v4h*)d)[i] = h;
}

// ---------------- 128x128 GEMM, C = A * W^T + bias --------------------------
// A: [M_,1024] f16 row-major.  W: [1024,1024] f16 in original (n-major) layout
// W[n][k], so B-fragments are contiguous.  mode 0: f16 out [B,H,T,hd];
// mode 1: f16 out transposed [B,H,hd,T] (for V);  mode 2: f32 out [M_,1024].
// Double-buffered LDS, software-pipelined; async global->LDS when available.
__global__ __launch_bounds__(256) void gemm128(
    const _Float16* __restrict__ A, const _Float16* __restrict__ W,
    const float* __restrict__ bias, _Float16* __restrict__ outh,
    float* __restrict__ outf, int mode)
{
  __shared__ _Float16 As[2][128 * 40];
  __shared__ _Float16 Bs[2][128 * 40];
  const int tid  = threadIdx.x;
  const int lane = tid & 31;
  const int w    = tid >> 5;
  const int wm   = w & 3;        // 4 waves along M (32 rows each)
  const int wn   = w >> 2;       // 2 waves along N (64 cols each)
  const int m0   = blockIdx.y * 128;
  const int n0   = blockIdx.x * 128;

  // per-thread staging coordinates: two b128 chunks per tile per matrix
  const int i0 = tid * 8,        r0 = i0 >> 5, c0 = i0 & 31;
  const int i1 = (tid + 256) * 8, r1 = i1 >> 5, c1 = i1 & 31;

#if defined(HAVE_ASYNC_LDS)
#define STAGE_LOAD(k0, buf)                                                 \
  do {                                                                      \
    ASYNC_CP_B128(&A[(size_t)(m0 + r0) * D_ + (k0) + c0], &As[buf][r0 * 40 + c0]); \
    ASYNC_CP_B128(&A[(size_t)(m0 + r1) * D_ + (k0) + c1], &As[buf][r1 * 40 + c1]); \
    ASYNC_CP_B128(&W[(size_t)(n0 + r0) * D_ + (k0) + c0], &Bs[buf][r0 * 40 + c0]); \
    ASYNC_CP_B128(&W[(size_t)(n0 + r1) * D_ + (k0) + c1], &Bs[buf][r1 * 40 + c1]); \
  } while (0)
#define STAGE_STORE(buf) ((void)0)
#define STAGE_FENCE()    WAIT_ASYNC()
#else
  uint4 ra0, ra1, rb0, rb1;
#define STAGE_LOAD(k0, buf)                                                 \
  do {                                                                      \
    ra0 = *(const uint4*)&A[(size_t)(m0 + r0) * D_ + (k0) + c0];            \
    ra1 = *(const uint4*)&A[(size_t)(m0 + r1) * D_ + (k0) + c1];            \
    rb0 = *(const uint4*)&W[(size_t)(n0 + r0) * D_ + (k0) + c0];            \
    rb1 = *(const uint4*)&W[(size_t)(n0 + r1) * D_ + (k0) + c1];            \
  } while (0)
#define STAGE_STORE(buf)                                                    \
  do {                                                                      \
    *(uint4*)&As[buf][r0 * 40 + c0] = ra0;                                  \
    *(uint4*)&As[buf][r1 * 40 + c1] = ra1;                                  \
    *(uint4*)&Bs[buf][r0 * 40 + c0] = rb0;                                  \
    *(uint4*)&Bs[buf][r1 * 40 + c1] = rb1;                                  \
  } while (0)
#define STAGE_FENCE() ((void)0)
#endif

  v8f acc[2][4] = {};

  // prologue: fill buffer 0
  STAGE_LOAD(0, 0);
  STAGE_STORE(0);
  STAGE_FENCE();
  __syncthreads();

  const int NIT = D_ / 32;
  for (int it = 0; it < NIT; ++it) {
    const int cur = it & 1, nxt = cur ^ 1;
    const bool has_next = (it + 1) < NIT;
    if (has_next) STAGE_LOAD((it + 1) * 32, nxt);   // in flight during compute

    v16h af[2], bf[4];
#pragma unroll
    for (int i = 0; i < 2; ++i) af[i] = frag_a_rowmajor(&As[cur][(wm * 32 + i * 16) * 40], 40);
#pragma unroll
    for (int j = 0; j < 4; ++j) bf[j] = frag_b_nmajor(&Bs[cur][(wn * 64 + j * 16) * 40], 40);
#pragma unroll
    for (int i = 0; i < 2; ++i)
#pragma unroll
      for (int j = 0; j < 4; ++j)
        acc[i][j] = __builtin_amdgcn_wmma_f32_16x16x32_f16(
            false, af[i], false, bf[j], (short)0, acc[i][j], false, false);

    if (has_next) STAGE_STORE(nxt);   // fallback path: loadcnt wait lands here
    STAGE_FENCE();                    // async path: s_wait_asynccnt 0
    __syncthreads();
  }

  const int row0 = (lane < 16) ? 0 : 8;
  const int ncol = lane & 15;
#pragma unroll
  for (int i = 0; i < 2; ++i)
#pragma unroll
    for (int j = 0; j < 4; ++j) {
      const int nb = n0 + wn * 64 + j * 16 + ncol;
      const float bv = bias[nb];
#pragma unroll
      for (int v = 0; v < 8; ++v) {
        const int m = m0 + wm * 32 + i * 16 + row0 + v;
        const float val = acc[i][j][v] + bv;
        if (mode == 2) {
          outf[(size_t)m * D_ + nb] = val;
        } else {
          const int b = m >> 11, t = m & (T_ - 1);
          const int h = nb >> 6, dh = nb & (HD_ - 1);
          size_t o = (mode == 1)
              ? (((size_t)(b * H_ + h) * HD_ + dh) * T_ + t)   // V: [bh][hd][T]
              : (((size_t)(b * H_ + h) * T_ + t) * HD_ + dh);  // Q,K: [bh][T][hd]
          outh[o] = (_Float16)val;
        }
      }
    }
#undef STAGE_LOAD
#undef STAGE_STORE
#undef STAGE_FENCE
}

// ---------------- flash attention with causal mask + linear decay -----------
// Q,K: [B*H][T][hd] f16,  Vt: [B*H][hd][T] f16,  O: [B][T][D] f16.
// score = (q.k)/8 - (i-j) for j<=i else -inf ; online softmax over 32-key tiles.
__global__ __launch_bounds__(128) void attn_kernel(
    const _Float16* __restrict__ Q, const _Float16* __restrict__ K,
    const _Float16* __restrict__ Vt, _Float16* __restrict__ O)
{
  __shared__ _Float16 pst_all[4][16 * 40];
  const int lane = threadIdx.x & 31;
  const int w    = threadIdx.x >> 5;
  _Float16* pst  = pst_all[w];
  const int bh = blockIdx.y;
  const int b  = bh >> 4, h = bh & 15;
  const int q0 = blockIdx.x * 64 + w * 16;
  const _Float16* Qp = Q  + (size_t)bh * T_ * HD_;
  const _Float16* Kp = K  + (size_t)bh * T_ * HD_;
  const _Float16* Vp = Vt + (size_t)bh * HD_ * T_;

  v16h qf[2];
#pragma unroll
  for (int c = 0; c < 2; ++c)
    qf[c] = frag_a_rowmajor(Qp + (size_t)q0 * HD_ + c * 32, HD_);

  v8f o[4] = {};
  float rm[8], rs[8];
#pragma unroll
  for (int v = 0; v < 8; ++v) { rm[v] = -1e30f; rs[v] = 0.0f; }

  const int row0 = (lane < 16) ? 0 : 8;
  const int ncol = lane & 15;
  const float SCALE = 0.125f;   // 1/sqrt(64)

  for (int j0 = 0; j0 < q0 + 16; j0 += 32) {
    // S = Q * K^T : two 16x16 score tiles (keys j0..j0+15, j0+16..j0+31)
    v8f s0 = {}, s1 = {};
#pragma unroll
    for (int c = 0; c < 2; ++c) {
      v16h kb0 = frag_b_nmajor(Kp + (size_t)j0 * HD_ + c * 32, HD_);
      v16h kb1 = frag_b_nmajor(Kp + (size_t)(j0 + 16) * HD_ + c * 32, HD_);
      s0 = __builtin_amdgcn_wmma_f32_16x16x32_f16(false, qf[c], false, kb0, (short)0, s0, false, false);
      s1 = __builtin_amdgcn_wmma_f32_16x16x32_f16(false, qf[c], false, kb1, (short)0, s1, false, false);
    }
    // mask + decay + online softmax (C-layout: row=(row0+v), col=ncol)
#pragma unroll
    for (int v = 0; v < 8; ++v) {
      const int i  = q0 + row0 + v;
      const int ja = j0 + ncol;
      const int jb = j0 + 16 + ncol;
      float sa = (ja <= i) ? (s0[v] * SCALE - (float)(i - ja)) : -1e30f;
      float sb = (jb <= i) ? (s1[v] * SCALE - (float)(i - jb)) : -1e30f;
      float tm = fmaxf(sa, sb);
#pragma unroll
      for (int m = 1; m < 16; m <<= 1) tm = fmaxf(tm, __shfl_xor(tm, m, 32));
      const float nm   = fmaxf(rm[v], tm);
      const float corr = __expf(rm[v] - nm);
      rm[v] = nm;
      const float pa = __expf(sa - nm);
      const float pb = __expf(sb - nm);
      float ps = pa + pb;
#pragma unroll
      for (int m = 1; m < 16; m <<= 1) ps += __shfl_xor(ps, m, 32);
      rs[v] = rs[v] * corr + ps;
#pragma unroll
      for (int j = 0; j < 4; ++j) o[j][v] *= corr;
      // stage P (C-layout -> LDS) for relayout into an A fragment
      pst[(row0 + v) * 40 + ncol]      = (_Float16)pa;
      pst[(row0 + v) * 40 + 16 + ncol] = (_Float16)pb;
    }
    asm volatile("s_wait_dscnt 0" ::: "memory");   // per-wave LDS RAW fence
    v16h pf = frag_a_rowmajor(pst, 40);            // P as 16x32 A fragment
    // O += P * V  (Vt is dim-major so the B fragment is contiguous)
#pragma unroll
    for (int j = 0; j < 4; ++j) {
      v16h vb = frag_b_nmajor(Vp + (size_t)(j * 16) * T_ + j0, T_);
      o[j] = __builtin_amdgcn_wmma_f32_16x16x32_f16(false, pf, false, vb, (short)0, o[j], false, false);
    }
  }

#pragma unroll
  for (int j = 0; j < 4; ++j) {
    const int col = h * HD_ + j * 16 + ncol;
#pragma unroll
    for (int v = 0; v < 8; ++v) {
      const int t = q0 + row0 + v;
      O[(size_t)(b * T_ + t) * D_ + col] = (_Float16)(o[j][v] / rs[v]);
    }
  }
}

// ---------------- launcher ---------------------------------------------------
extern "C" void kernel_launch(void* const* d_in, const int* in_sizes, int n_in,
                              void* d_out, int out_size, void* d_ws, size_t ws_size,
                              hipStream_t stream) {
  (void)in_sizes; (void)n_in; (void)out_size; (void)ws_size;
  const float* x  = (const float*)d_in[0];
  const float* Wq = (const float*)d_in[1];
  const float* bq = (const float*)d_in[2];
  const float* Wk = (const float*)d_in[3];
  const float* bk = (const float*)d_in[4];
  const float* Wv = (const float*)d_in[5];
  const float* bv = (const float*)d_in[6];
  const float* Wo = (const float*)d_in[7];
  const float* bo = (const float*)d_in[8];

  char* ws = (char*)d_ws;
  const size_t MB = (size_t)1 << 20;
  _Float16* xh  = (_Float16*)(ws + 0 * MB);   // 8 MiB  [M_,D_]
  _Float16* Wqh = (_Float16*)(ws + 8 * MB);   // 2 MiB each, n-major
  _Float16* Wkh = (_Float16*)(ws + 10 * MB);
  _Float16* Wvh = (_Float16*)(ws + 12 * MB);
  _Float16* Woh = (_Float16*)(ws + 14 * MB);
  _Float16* Qb  = (_Float16*)(ws + 16 * MB);  // 8 MiB [bh][T][hd]
  _Float16* Kb  = (_Float16*)(ws + 24 * MB);  // 8 MiB [bh][T][hd]
  _Float16* Vb  = (_Float16*)(ws + 32 * MB);  // 8 MiB [bh][hd][T]
  _Float16* Ob  = (_Float16*)(ws + 40 * MB);  // 8 MiB [B,T,D]

  const int nx4 = M_ * D_ / 4, nw4 = D_ * D_ / 4;
  cvt_f32_f16<<<nx4 / 256, 256, 0, stream>>>(x,  xh,  nx4);
  cvt_f32_f16<<<nw4 / 256, 256, 0, stream>>>(Wq, Wqh, nw4);
  cvt_f32_f16<<<nw4 / 256, 256, 0, stream>>>(Wk, Wkh, nw4);
  cvt_f32_f16<<<nw4 / 256, 256, 0, stream>>>(Wv, Wvh, nw4);
  cvt_f32_f16<<<nw4 / 256, 256, 0, stream>>>(Wo, Woh, nw4);

  dim3 gg(D_ / 128, M_ / 128);
  gemm128<<<gg, 256, 0, stream>>>(xh, Wqh, bq, Qb, nullptr, 0);
  gemm128<<<gg, 256, 0, stream>>>(xh, Wkh, bk, Kb, nullptr, 0);
  gemm128<<<gg, 256, 0, stream>>>(xh, Wvh, bv, Vb, nullptr, 1);

  attn_kernel<<<dim3(T_ / 64, B_ * H_), 128, 0, stream>>>(Qb, Kb, Vb, Ob);

  gemm128<<<gg, 256, 0, stream>>>(Ob, Woh, bo, nullptr, (float*)d_out, 2);
}
